// Model_52518860096427
// MI455X (gfx1250) — matrix operand order repaired and checked by reading, not compiled, not run on hardware
//
#include <hip/hip_runtime.h>

typedef __attribute__((ext_vector_type(16))) _Float16 v16h;
typedef __attribute__((ext_vector_type(8)))  _Float16 v8h;
typedef __attribute__((ext_vector_type(4)))  _Float16 v4h;
typedef __attribute__((ext_vector_type(2)))  _Float16 v2h;
typedef __attribute__((ext_vector_type(8)))  float    v8f;
typedef __attribute__((ext_vector_type(4)))  float    v4f;
typedef __attribute__((ext_vector_type(8)))  unsigned v8u;

#define B_ 2
#define H_ 16
#define S_ 2048
#define D_ 128

#define QBLK   128   // query rows per workgroup (8 waves x 16)
#define KCHUNK 32    // keys per inner iteration
#define NWAVES 8

#define KS_STRIDE 136  // halves (272B rows)
#define VT_STRIDE 40   // halves (80B rows)
#define OS_STRIDE 132  // floats (528B rows)

#define CAT16(l8, h8) __builtin_shufflevector((l8), (h8), 0,1,2,3,4,5,6,7,8,9,10,11,12,13,14,15)
#define WMMA_F16(a, b, c) \
    __builtin_amdgcn_wmma_f32_16x16x32_f16(false, (a), false, (b), (short)0, (c), false, false)

static __device__ __forceinline__ unsigned pk2(float a, float b) {
    v2h h; h[0] = (_Float16)a; h[1] = (_Float16)b;
    return __builtin_bit_cast(unsigned, h);
}

__global__ __launch_bounds__(256)
void fa_fwd_t(const float* __restrict__ Q,
              const float* __restrict__ K,
              const float* __restrict__ V,
              float* __restrict__ O)
{
    // double-buffered K (key-major) and V (d-major/transposed) f16 tiles
    __shared__ _Float16 Ks[2][KCHUNK][KS_STRIDE];
    __shared__ _Float16 Vt[2][D_][VT_STRIDE];
    __shared__ float    Os[NWAVES][16][OS_STRIDE];   // epilogue transpose pad

    const int tid  = threadIdx.x;
    const int wave = tid >> 5;
    const int lane = tid & 31;
    const int hi   = lane >> 4;     // half-wave
    const int lo   = lane & 15;     // lane-in-half == query column

    const int    bh    = blockIdx.y;
    const size_t base  = (size_t)bh * S_ * D_;
    const int    qrow0 = blockIdx.x * QBLK + wave * 16;

    // ---------------- Q -> registers as WMMA B-fragments (Q^T, d x 16q) ----------------
    // B fragment in this code's contiguous k order (same as the K A-fragment): lane col N = lo (query), element j <-> Kdim = 16*hi + j
    v16h qfrag[4];
    {
        const float* qp = Q + base + (size_t)(qrow0 + lo) * D_;
        #pragma unroll
        for (int kc = 0; kc < 4; ++kc) {
            const int db = kc * 32 + 16 * hi;
            v4f a = *(const v4f*)(qp + db);
            v4f b = *(const v4f*)(qp + db + 4);
            v4f c = *(const v4f*)(qp + db + 8);
            v4f d = *(const v4f*)(qp + db + 12);
            #pragma unroll
            for (int j = 0; j < 4; ++j) {
                qfrag[kc][j]      = (_Float16)a[j];
                qfrag[kc][4 + j]  = (_Float16)b[j];
                qfrag[kc][8 + j]  = (_Float16)c[j];
                qfrag[kc][12 + j] = (_Float16)d[j];
            }
        }
    }

    // O^T accumulators: 8 d-tiles; lane = query, VGPR row = d (r + 8*hi) within tile
    v8f accT[8];
    #pragma unroll
    for (int dt = 0; dt < 8; ++dt) accT[dt] = (v8f){};
    float m_run = -3.0e38f, l_run = 0.0f;

    // staging: each thread owns a 4key x 4d block; fully coalesced global b128 loads
    const int dblk = (tid & 31) * 4;   // 0..124
    const int kblk = (tid >> 5) * 4;   // 0..28

    const int NKB = S_ / KCHUNK;
    v4f kreg[4], vreg[4];
    {
        const float* kp = K + base + (size_t)kblk * D_ + dblk;
        const float* vp = V + base + (size_t)kblk * D_ + dblk;
        #pragma unroll
        for (int r = 0; r < 4; ++r) {
            kreg[r] = *(const v4f*)(kp + (size_t)r * D_);
            vreg[r] = *(const v4f*)(vp + (size_t)r * D_);
        }
    }

    int buf = 0;
    for (int kb = 0; kb < NKB; ++kb) {
        // ---- store staged registers into LDS[buf] (f16) ----
        #pragma unroll
        for (int r = 0; r < 4; ++r) {   // K row-major
            v4h kk;
            kk[0] = (_Float16)kreg[r][0]; kk[1] = (_Float16)kreg[r][1];
            kk[2] = (_Float16)kreg[r][2]; kk[3] = (_Float16)kreg[r][3];
            *(v4h*)&Ks[buf][kblk + r][dblk] = kk;
        }
        #pragma unroll
        for (int j = 0; j < 4; ++j) {   // V transposed (d-major)
            v4h vv;
            vv[0] = (_Float16)vreg[0][j]; vv[1] = (_Float16)vreg[1][j];
            vv[2] = (_Float16)vreg[2][j]; vv[3] = (_Float16)vreg[3][j];
            *(v4h*)&Vt[buf][dblk + j][kblk] = vv;
        }
        __syncthreads();

        // ---- kick off next chunk's global loads (overlap with WMMA below) ----
        if (kb + 1 < NKB) {
            const size_t roff = base + (size_t)((kb + 1) * KCHUNK + kblk) * D_ + dblk;
            #pragma unroll
            for (int r = 0; r < 4; ++r) {
                kreg[r] = *(const v4f*)(K + roff + (size_t)r * D_);
                vreg[r] = *(const v4f*)(V + roff + (size_t)r * D_);
            }
        }
        if (kb + 2 < NKB) {
            const size_t poff = base + (size_t)((kb + 2) * KCHUNK + kblk) * D_ + dblk;
            __builtin_prefetch(K + poff, 0, 1);   // global_prefetch_b8
            __builtin_prefetch(V + poff, 0, 1);
        }

        // ---- S^T = K_chunk x Q^T : two 16x16 tiles (keys 0-15 / 16-31 as rows) ----
        // A fragment in this code's contiguous k order (same as qfrag): lane row = key, elem j -> Kdim 16*hi+j
        v8f st0 = (v8f){}, st1 = (v8f){};
        #pragma unroll
        for (int kc = 0; kc < 4; ++kc) {
            const int c1 = kc * 32 + 16 * hi;
            const int c2 = kc * 32 + 16 * hi + 8;
            v8h a00 = *(const v8h*)&Ks[buf][lo][c1];
            v8h a01 = *(const v8h*)&Ks[buf][lo][c2];
            v8h a10 = *(const v8h*)&Ks[buf][16 + lo][c1];
            v8h a11 = *(const v8h*)&Ks[buf][16 + lo][c2];
            st0 = WMMA_F16(CAT16(a00, a01), qfrag[kc], st0);
            st1 = WMMA_F16(CAT16(a10, a11), qfrag[kc], st1);
        }
        // lane (lo,hi) now holds, for query lo: st0[r] = s[key r+8*hi], st1[r] = s[key 16+r+8*hi]

        // ---- online softmax: per-lane VALU tree + ONE cross-half shuffle each ----
        float mx = fmaxf(st0[0], st1[0]);
        #pragma unroll
        for (int r = 1; r < 8; ++r) mx = fmaxf(mx, fmaxf(st0[r], st1[r]));
        mx = fmaxf(mx, __shfl_xor(mx, 16, 32));
        const float mnew  = fmaxf(m_run, mx);
        const float scale = __expf(m_run - mnew);

        v8f e0, e1;
        float rs = 0.0f;
        #pragma unroll
        for (int r = 0; r < 8; ++r) {
            e0[r] = __expf(st0[r] - mnew);
            e1[r] = __expf(st1[r] - mnew);
            rs += e0[r] + e1[r];
        }
        rs += __shfl_xor(rs, 16, 32);
        l_run = l_run * scale + rs;
        m_run = mnew;

        #pragma unroll
        for (int dt = 0; dt < 8; ++dt)
            #pragma unroll
            for (int r = 0; r < 8; ++r)
                accT[dt][r] *= scale;

        // ---- build P^T B-fragment in registers (no LDS round-trip) ----
        // B elem j <-> key = 16*hi + j.  hi=0 needs keys 0-7(own e0) + 8-15(partner e0)
        //                               hi=1 needs keys 16-23(partner e1) + 24-31(own e1)
        unsigned o0[4], o1[4], x0[4], x1[4];
        #pragma unroll
        for (int i = 0; i < 4; ++i) {
            o0[i] = pk2(e0[2 * i], e0[2 * i + 1]);
            o1[i] = pk2(e1[2 * i], e1[2 * i + 1]);
        }
        #pragma unroll
        for (int i = 0; i < 4; ++i) {
            x0[i] = __shfl_xor(o0[i], 16, 32);
            x1[i] = __shfl_xor(o1[i], 16, 32);
        }
        v8u pu;
        #pragma unroll
        for (int i = 0; i < 4; ++i) {
            pu[i]     = hi ? x1[i] : o0[i];
            pu[4 + i] = hi ? o1[i] : x0[i];
        }
        const v16h pt = __builtin_bit_cast(v16h, pu);

        // ---- O^T += V^T x P^T : 8 d-tiles, K = 32 keys ----
        #pragma unroll
        for (int dt = 0; dt < 8; ++dt) {
            v8h w0 = *(const v8h*)&Vt[buf][dt * 16 + lo][16 * hi];
            v8h w1 = *(const v8h*)&Vt[buf][dt * 16 + lo][16 * hi + 8];
            accT[dt] = WMMA_F16(CAT16(w0, w1), pt, accT[dt]);
        }

        buf ^= 1;
    }

    // ---------------- epilogue: normalize, transpose via LDS, coalesced store ----------------
    const float inv = 1.0f / l_run;
    #pragma unroll
    for (int dt = 0; dt < 8; ++dt) {
        v4f y0, y1;
        #pragma unroll
        for (int r = 0; r < 4; ++r) {
            y0[r] = accT[dt][r] * inv;
            y1[r] = accT[dt][4 + r] * inv;
        }
        *(v4f*)&Os[wave][lo][dt * 16 + 8 * hi]     = y0;   // d = dt*16 + 8*hi + r
        *(v4f*)&Os[wave][lo][dt * 16 + 8 * hi + 4] = y1;
    }

    float* op = O + base + (size_t)qrow0 * D_;
    #pragma unroll
    for (int q = 0; q < 16; ++q) {
        v4f t = *(const v4f*)&Os[wave][q][lane * 4];
        *(v4f*)(op + (size_t)q * D_ + lane * 4) = t;       // 512B contiguous per wave
    }
}

extern "C" void kernel_launch(void* const* d_in, const int* in_sizes, int n_in,
                              void* d_out, int out_size, void* d_ws, size_t ws_size,
                              hipStream_t stream) {
    const float* Q = (const float*)d_in[0];
    const float* K = (const float*)d_in[1];
    const float* V = (const float*)d_in[2];
    float*       O = (float*)d_out;
    (void)in_sizes; (void)n_in; (void)out_size; (void)d_ws; (void)ws_size;

    dim3 grid(S_ / QBLK, B_ * H_);
    fa_fwd_t<<<grid, 256, 0, stream>>>(Q, K, V, O);
}
